// Head_12378095747778
// MI455X (gfx1250) — matrix operand order repaired and checked by reading, not compiled, not run on hardware
//
#include <hip/hip_runtime.h>
#include <stdint.h>
#include <math.h>

// ---------------------------------------------------------------------------
// Causal single-head attention for MI455X (gfx1250), bf16 WMMA pipeline.
//   B=8, T=2048, C=768 (n_embed), HS=256 (head size)
// Software-pipelined: operand loads for chunk i+1 overlap WMMAs of chunk i.
// ---------------------------------------------------------------------------

#define BATCH 8
#define TSEQ  2048
#define CEMB  768
#define HSZ   256

typedef __attribute__((ext_vector_type(16))) __bf16 v16bf;
typedef __attribute__((ext_vector_type(8)))  float  v8f;

union ABf { uint32_t u[8]; v16bf v; };

__device__ __forceinline__ v8f wmma_bf16(const ABf& a, const ABf& b, v8f c) {
  // D = A(16x32 bf16) * B(32x16 bf16) + C(16x16 f32)
  return __builtin_amdgcn_wmma_f32_16x16x32_bf16(
      /*neg_a=*/false, a.v, /*neg_b=*/false, b.v,
      /*c_mod=*/(short)0, c, /*reuse_a=*/false, /*reuse_b=*/false);
}

__device__ __forceinline__ uint16_t f32_to_bf16(float f) {
  union { float f; uint32_t u; } x; x.f = f;
  uint32_t u = x.u;
  u += 0x7FFFu + ((u >> 16) & 1u);      // round-to-nearest-even
  return (uint16_t)(u >> 16);
}

// ---------------------------------------------------------------------------
// fp32 -> bf16 elementwise
// ---------------------------------------------------------------------------
__global__ void cvt_bf16_kernel(const float* __restrict__ src,
                                uint16_t* __restrict__ dst, int n) {
  int i = blockIdx.x * blockDim.x + threadIdx.x;
  int stride = gridDim.x * blockDim.x;
  for (; i < n; i += stride) dst[i] = f32_to_bf16(src[i]);
}

// fp32 [rows, cols] -> bf16 transposed [cols, rows]
__global__ void cvt_bf16_t_kernel(const float* __restrict__ src,
                                  uint16_t* __restrict__ dst,
                                  int rows, int cols) {
  int n = rows * cols;
  int i = blockIdx.x * blockDim.x + threadIdx.x;
  int stride = gridDim.x * blockDim.x;
  for (; i < n; i += stride) {
    int r = i / cols, c = i - r * cols;
    dst[(size_t)c * rows + r] = f32_to_bf16(src[i]);
  }
}

// ---------------------------------------------------------------------------
// Projection: out[m,n] = scale * sum_k x[m,k] * Wt[n,k]
//   xbf : [B*T, C] bf16 row-major
//   wt  : [HS, C]  bf16 (W transposed)
//   out : row-major bf16 [B*T, HS]           (transpose_out == 0)
//         per-batch transposed bf16 [B,HS,T] (transpose_out == 1)
// One wave computes a 16(M) x 64(N) strip; 8 waves per block.
// Double-buffered operand registers: loads(i+1) overlap wmma(i).
// ---------------------------------------------------------------------------
__global__ __launch_bounds__(256) void proj_kernel(
    const uint16_t* __restrict__ xbf,
    const uint16_t* __restrict__ wt,
    uint16_t* __restrict__ out,
    float scale, int transpose_out) {
  const int lane = threadIdx.x & 31;
  const int half = lane >> 4;
  const int lm   = lane & 15;
  const int wave_global = blockIdx.x * (blockDim.x >> 5) + (threadIdx.x >> 5);
  const int mtile  = wave_global >> 2;        // (B*T/16) row tiles
  const int nstrip = wave_global & 3;         // HS/64 strips
  const int m0 = mtile * 16;
  const int n0 = nstrip * 64;

  v8f acc0 = {}, acc1 = {}, acc2 = {}, acc3 = {};

  const uint16_t* xrow = xbf + (size_t)(m0 + lm) * CEMB;
  const uint16_t* w0 = wt + (size_t)(n0 + lm) * CEMB;
  const uint16_t* w1 = w0 + (size_t)16 * CEMB;
  const uint16_t* w2 = w0 + (size_t)32 * CEMB;
  const uint16_t* w3 = w0 + (size_t)48 * CEMB;

  auto load_chunk = [&](int kb, ABf& a, ABf& b0, ABf& b1, ABf& b2, ABf& b3) {
#pragma unroll
    for (int j = 0; j < 8; ++j) {
      int ka = kb + 2 * j + 16 * half;
      int kbk = kb + 2 * j + 16 * half;
      a.u[j]  = *(const uint32_t*)(xrow + ka);
      b0.u[j] = *(const uint32_t*)(w0 + kbk);
      b1.u[j] = *(const uint32_t*)(w1 + kbk);
      b2.u[j] = *(const uint32_t*)(w2 + kbk);
      b3.u[j] = *(const uint32_t*)(w3 + kbk);
    }
  };
  auto step = [&](const ABf& a, const ABf& b0, const ABf& b1,
                  const ABf& b2, const ABf& b3) {
    acc0 = wmma_bf16(a, b0, acc0);
    acc1 = wmma_bf16(a, b1, acc1);
    acc2 = wmma_bf16(a, b2, acc2);
    acc3 = wmma_bf16(a, b3, acc3);
  };

  ABf aA, b0A, b1A, b2A, b3A;   // buffer A
  ABf aB, b0B, b1B, b2B, b3B;   // buffer B
  load_chunk(0, aA, b0A, b1A, b2A, b3A);
  // 24 K-steps total; unrolled x2 so buffer selection is static
  for (int c = 0; c < 22; c += 2) {
    load_chunk((c + 1) * 32, aB, b0B, b1B, b2B, b3B);
    step(aA, b0A, b1A, b2A, b3A);
    load_chunk((c + 2) * 32, aA, b0A, b1A, b2A, b3A);
    step(aB, b0B, b1B, b2B, b3B);
  }
  load_chunk(23 * 32, aB, b0B, b1B, b2B, b3B);
  step(aA, b0A, b1A, b2A, b3A);
  step(aB, b0B, b1B, b2B, b3B);

  v8f accs[4] = { acc0, acc1, acc2, acc3 };
  if (!transpose_out) {
#pragma unroll
    for (int nt = 0; nt < 4; ++nt) {
      v8f acc = accs[nt];
#pragma unroll
      for (int r = 0; r < 8; ++r) {
        int row = m0 + r + 8 * half;        // C/D layout: M = r + 8*half
        out[(size_t)row * HSZ + n0 + nt * 16 + lm] = f32_to_bf16(acc[r] * scale);
      }
    }
  } else {
    // per-batch transposed store: Vt[b][hs][t]; 8 consecutive t per lane -> 16B store
    const int bb = m0 / TSEQ;
    const int tstart = (m0 % TSEQ) + 8 * half;
#pragma unroll
    for (int nt = 0; nt < 4; ++nt) {
      v8f acc = accs[nt];
      uint32_t pk[4];
#pragma unroll
      for (int i2 = 0; i2 < 4; ++i2)
        pk[i2] = (uint32_t)f32_to_bf16(acc[2 * i2] * scale) |
                 ((uint32_t)f32_to_bf16(acc[2 * i2 + 1] * scale) << 16);
      int col = n0 + nt * 16 + lm;
      uint16_t* dstp = out + (size_t)bb * HSZ * TSEQ + (size_t)col * TSEQ + tstart;
      uint4 v; v.x = pk[0]; v.y = pk[1]; v.z = pk[2]; v.w = pk[3];
      *(uint4*)dstp = v;
    }
  }
}

// ---------------------------------------------------------------------------
// Flash attention (causal). 4 waves/block, 16 query rows per wave.
//   Qbf : [B,T,HS] bf16, pre-scaled by C^-0.5
//   Kbf : [B,T,HS] bf16
//   Vt  : [B,HS,T] bf16 (transposed)
//   outp: [B,T,HS] fp32
// ---------------------------------------------------------------------------
__global__ __launch_bounds__(128) void attn_kernel(
    const uint16_t* __restrict__ Qbf,
    const uint16_t* __restrict__ Kbf,
    const uint16_t* __restrict__ Vt,
    float* __restrict__ outp) {
  __shared__ uint16_t p_lds[4 * 16 * 32];   // per-wave 16x32 bf16 P tile

  const int wave = threadIdx.x >> 5;
  const int lane = threadIdx.x & 31;
  const int half = lane >> 4;
  const int lm   = lane & 15;

  const int b    = blockIdx.x / (TSEQ / 64);
  const int qblk = blockIdx.x % (TSEQ / 64);
  const int q0   = qblk * 64 + wave * 16;

  const uint16_t* Qb = Qbf + (size_t)b * TSEQ * HSZ;
  const uint16_t* Kb = Kbf + (size_t)b * TSEQ * HSZ;
  const uint16_t* Vb = Vt  + (size_t)b * HSZ * TSEQ;
  uint16_t* pl = p_lds + wave * (16 * 32);

  // Q A-operands: 8 chunks over HS=256, resident for the whole key loop
  ABf qa[8];
  const uint16_t* qrow = Qb + (size_t)(q0 + lm) * HSZ;
#pragma unroll
  for (int c = 0; c < 8; ++c) {
#pragma unroll
    for (int j = 0; j < 8; ++j) {
      int k = c * 32 + 2 * j + 16 * half;
      qa[c].u[j] = *(const uint32_t*)(qrow + k);
    }
  }

  v8f o[16];
#pragma unroll
  for (int t = 0; t < 16; ++t) o[t] = (v8f){};
  float mrow[8], lrow[8];
#pragma unroll
  for (int r = 0; r < 8; ++r) { mrow[r] = -3.0e38f; lrow[r] = 0.0f; }

  const float L2E = 1.44269504f;
  const float NEG = -3.0e38f;
  const int nkb = (q0 + 16 + 31) / 32;     // 32-key blocks under causal mask

  for (int kb = 0; kb < nkb; ++kb) {
    const int k0 = kb * 32;

    auto loadK = [&](int c, ABf& t0, ABf& t1) {
      const uint16_t* kr0 = Kb + (size_t)(k0 + lm) * HSZ + c * 32;
      const uint16_t* kr1 = Kb + (size_t)(k0 + 16 + lm) * HSZ + c * 32;
#pragma unroll
      for (int j = 0; j < 8; ++j) {
        t0.u[j] = *(const uint32_t*)(kr0 + 2 * j + 16 * half);
        t1.u[j] = *(const uint32_t*)(kr1 + 2 * j + 16 * half);
      }
    };
    auto loadV = [&](int t, ABf& dst) {
      const uint16_t* vr = Vb + (size_t)(t * 16 + lm) * TSEQ + k0;
#pragma unroll
      for (int j = 0; j < 8; ++j)
        dst.u[j] = *(const uint32_t*)(vr + 2 * j + 16 * half);
    };

    // ---- S = Q @ K^T : two 16x16 tiles, K-chunk loads pipelined ----
    v8f s0 = {}, s1 = {};
    ABf kt0[2], kt1[2];
    loadK(0, kt0[0], kt1[0]);
#pragma unroll
    for (int c = 0; c < 8; ++c) {
      if (c < 7) loadK(c + 1, kt0[(c + 1) & 1], kt1[(c + 1) & 1]);
      s0 = wmma_bf16(qa[c], kt0[c & 1], s0);
      s1 = wmma_bf16(qa[c], kt1[c & 1], s1);
    }

    // ---- causal mask (scale already folded into Q) ----
#pragma unroll
    for (int r = 0; r < 8; ++r) {
      int q = q0 + r + 8 * half;
      s0[r] = (k0 + lm      > q) ? NEG : s0[r];
      s1[r] = (k0 + 16 + lm > q) ? NEG : s1[r];
    }

    // issue first V tile load now: overlaps the softmax VALU work below
    ABf vt[2];
    loadV(0, vt[0]);

    // ---- online softmax (row stats live in 16-lane halves) ----
    float cexp[8];
#pragma unroll
    for (int r = 0; r < 8; ++r) {
      float mx = fmaxf(s0[r], s1[r]);
#pragma unroll
      for (int m = 1; m < 16; m <<= 1) mx = fmaxf(mx, __shfl_xor(mx, m, 32));
      float mnew = fmaxf(mrow[r], mx);
      float corr = __builtin_exp2f((mrow[r] - mnew) * L2E);
      float p0 = __builtin_exp2f((s0[r] - mnew) * L2E);
      float p1 = __builtin_exp2f((s1[r] - mnew) * L2E);
      float rs = p0 + p1;
#pragma unroll
      for (int m = 1; m < 16; m <<= 1) rs += __shfl_xor(rs, m, 32);
      lrow[r] = lrow[r] * corr + rs;
      mrow[r] = mnew;
      cexp[r] = corr;
      s0[r] = p0; s1[r] = p1;
    }

    // rescale O accumulators
#pragma unroll
    for (int t = 0; t < 16; ++t) {
#pragma unroll
      for (int r = 0; r < 8; ++r) o[t][r] *= cexp[r];
    }

    // ---- P (C/D layout) -> LDS -> A-operand layout ----
#pragma unroll
    for (int r = 0; r < 8; ++r) {
      int M = r + 8 * half;
      pl[M * 32 + lm]      = f32_to_bf16(s0[r]);
      pl[M * 32 + 16 + lm] = f32_to_bf16(s1[r]);
    }
    asm volatile("s_wait_dscnt 0" ::: "memory");   // same-wave LDS RAW
    ABf pa;
#pragma unroll
    for (int j = 0; j < 8; ++j) {
      int kk = 2 * j + 16 * half;
      pa.u[j] = *(const uint32_t*)(pl + lm * 32 + kk);
    }

    // ---- O += P @ V over all HS=256 columns, V loads pipelined ----
#pragma unroll
    for (int t = 0; t < 16; ++t) {
      if (t < 15) loadV(t + 1, vt[(t + 1) & 1]);
      o[t] = wmma_bf16(pa, vt[t & 1], o[t]);
    }
  }

  // ---- epilogue: normalize and store fp32 ----
  float* Ob = outp + (size_t)b * TSEQ * HSZ;
  float inv[8];
#pragma unroll
  for (int r = 0; r < 8; ++r) inv[r] = 1.0f / lrow[r];
#pragma unroll
  for (int t = 0; t < 16; ++t) {
#pragma unroll
    for (int r = 0; r < 8; ++r) {
      int row = q0 + r + 8 * half;
      Ob[(size_t)row * HSZ + t * 16 + lm] = o[t][r] * inv[r];
    }
  }
}

// ---------------------------------------------------------------------------
extern "C" void kernel_launch(void* const* d_in, const int* in_sizes, int n_in,
                              void* d_out, int out_size, void* d_ws, size_t ws_size,
                              hipStream_t stream) {
  (void)in_sizes; (void)n_in; (void)out_size; (void)ws_size;
  const float* x  = (const float*)d_in[0];
  const float* Wq = (const float*)d_in[1];
  const float* Wk = (const float*)d_in[2];
  const float* Wv = (const float*)d_in[3];
  float* outp = (float*)d_out;

  const size_t BT = (size_t)BATCH * TSEQ;

  // workspace layout (all 16B-aligned): ~51.6 MB total
  uint16_t* xbf = (uint16_t*)d_ws;                 // [BT, C]
  uint16_t* wqt = xbf + BT * CEMB;                 // [HS, C]
  uint16_t* wkt = wqt + (size_t)HSZ * CEMB;
  uint16_t* wvt = wkt + (size_t)HSZ * CEMB;
  uint16_t* Qb  = wvt + (size_t)HSZ * CEMB;        // [BT, HS]
  uint16_t* Kb  = Qb + BT * HSZ;                   // [BT, HS]
  uint16_t* Vt  = Kb + BT * HSZ;                   // [B, HS, T]

  cvt_bf16_kernel<<<2048, 256, 0, stream>>>(x, xbf, (int)(BT * CEMB));
  cvt_bf16_t_kernel<<<192, 256, 0, stream>>>(Wq, wqt, CEMB, HSZ);
  cvt_bf16_t_kernel<<<192, 256, 0, stream>>>(Wk, wkt, CEMB, HSZ);
  cvt_bf16_t_kernel<<<192, 256, 0, stream>>>(Wv, wvt, CEMB, HSZ);

  const float qscale = 1.0f / sqrtf((float)CEMB);  // reference scales by C^-0.5
  // waves = (BT/16) * (HS/64) = 4096 ; 8 waves/block -> 512 blocks
  proj_kernel<<<512, 256, 0, stream>>>(xbf, wqt, Qb, qscale, 0);
  proj_kernel<<<512, 256, 0, stream>>>(xbf, wkt, Kb, 1.0f, 0);
  proj_kernel<<<512, 256, 0, stream>>>(xbf, wvt, Vt, 1.0f, 1);

  attn_kernel<<<BATCH * (TSEQ / 64), 128, 0, stream>>>(Qb, Kb, Vt, outp);
}
